// HierarchicalReversibleAttention_49117245997547
// MI455X (gfx1250) — matrix operand order repaired and checked by reading, not compiled, not run on hardware
//
#include <hip/hip_runtime.h>
#include <hip/hip_bf16.h>

// ---------------------------------------------------------------------------
// MI455X (gfx1250): all GEMMs + attention matmuls on v_wmma_f32_16x16x32_bf16
// (f32 accumulate); flash-attention K/V tiles staged via async global->LDS
// (global_load_async_to_lds_b128 + s_wait_asynccnt), double buffered.
// ---------------------------------------------------------------------------

typedef __attribute__((ext_vector_type(16))) __bf16 v16bf;
typedef __attribute__((ext_vector_type(8)))  float  v8f;

union BfFrag { unsigned int u[8]; v16bf v; };

__device__ __forceinline__ unsigned short f2bf(float f) {
  unsigned int u = __builtin_bit_cast(unsigned int, f);
  u += 0x7fffu + ((u >> 16) & 1u);   // round-to-nearest-even
  return (unsigned short)(u >> 16);
}

__device__ __forceinline__ v8f wmma_bf16(const BfFrag& a, const BfFrag& b, v8f c) {
  return __builtin_amdgcn_wmma_f32_16x16x32_bf16(false, a.v, false, b.v,
                                                 (short)0, c, false, false);
}

// async 16B global -> LDS copy (per lane); LDS offset = low 32 bits of the
// generic pointer (flat LDS aperture truncation), tracked by ASYNCcnt.
__device__ __forceinline__ void async_copy_b128(const void* gptr, void* lptr) {
  unsigned int loff = (unsigned int)(unsigned long long)lptr;
  unsigned long long gaddr = (unsigned long long)gptr;
  asm volatile("global_load_async_to_lds_b128 %0, %1, off"
               :: "v"(loff), "v"(gaddr) : "memory");
}

// ---------------------------------------------------------------------------
// Generic GEMM: C[M,N] = A[M,K] * W[N,K]^T + bias[N]
// 256 threads (8 waves) per 64x64 tile; each wave owns a 16x32 C strip.
// LDS stages f32->bf16; next K-slab prefetched into L2 while computing.
// Requires M%64==0, N%64==0, K%32==0 (guaranteed by launch shapes).
// ---------------------------------------------------------------------------
__global__ void gemm_bf16_wmma(const float* __restrict__ A, int lda,
                               const float* __restrict__ W, int ldw,
                               const float* __restrict__ bias,
                               float* __restrict__ Cf,
                               unsigned short* __restrict__ Cb,
                               int ldc, int M, int N, int K)
{
  __shared__ unsigned short As[64 * 32];
  __shared__ unsigned short Ws[64 * 32];

  const int tid  = threadIdx.x;
  const int lane = tid & 31;
  const int wv   = tid >> 5;
  const int m0   = blockIdx.y * 64;
  const int n0   = blockIdx.x * 64;
  const int mt   = wv >> 1;           // M sub-tile 0..3
  const int np   = (wv & 1) * 2;      // N sub-tile pair base (0 or 2)
  const int hi   = lane >> 4;
  const int nl   = lane & 15;

  v8f acc0, acc1;
#pragma unroll
  for (int v = 0; v < 8; ++v) { acc0[v] = 0.f; acc1[v] = 0.f; }

  for (int k0 = 0; k0 < K; k0 += 32) {
    if (k0 + 32 < K) {               // L2 prefetch of next K slab
      int r = tid >> 2, c = (tid & 3) * 8;
      __builtin_prefetch(A + (size_t)(m0 + r) * lda + k0 + 32 + c, 0, 0);
      __builtin_prefetch(W + (size_t)(n0 + r) * ldw + k0 + 32 + c, 0, 0);
    }
#pragma unroll
    for (int i = 0; i < 8; ++i) {
      int idx = tid + i * 256;                 // 0..2047 (== r*32+c)
      int r = idx >> 5, c = idx & 31;
      As[idx] = f2bf(A[(size_t)(m0 + r) * lda + k0 + c]);
      Ws[idx] = f2bf(W[(size_t)(n0 + r) * ldw + k0 + c]);
    }
    __syncthreads();

    BfFrag af, bf0, bf1;
    const int ra = mt * 16 + nl;
#pragma unroll
    for (int v = 0; v < 8; ++v) {              // 16-bit A 16x32 layout
      int g = v >> 2, v2 = v & 3;
      int k = g * 16 + hi * 8 + v2 * 2;
      af.u[v] = *(const unsigned int*)&As[ra * 32 + k];
    }
    const int na = (np * 16) + nl;
    const int nb = ((np + 1) * 16) + nl;
#pragma unroll
    for (int v = 0; v < 8; ++v) {              // 16-bit B 32x16 layout
      int k = (v >> 2) * 16 + hi * 8 + (v & 3) * 2;
      bf0.u[v] = *(const unsigned int*)&Ws[na * 32 + k];
      bf1.u[v] = *(const unsigned int*)&Ws[nb * 32 + k];
    }
    acc0 = wmma_bf16(af, bf0, acc0);
    acc1 = wmma_bf16(af, bf1, acc1);
    __syncthreads();
  }

  const int c0 = n0 + np * 16 + nl;
  const int c1 = c0 + 16;
  const float b0 = bias ? bias[c0] : 0.f;
  const float b1 = bias ? bias[c1] : 0.f;
#pragma unroll
  for (int v = 0; v < 8; ++v) {
    int m = m0 + mt * 16 + hi * 8 + v;
    float x0 = acc0[v] + b0, x1 = acc1[v] + b1;
    if (Cb) {
      Cb[(size_t)m * ldc + c0] = f2bf(x0);
      Cb[(size_t)m * ldc + c1] = f2bf(x1);
    } else {
      Cf[(size_t)m * ldc + c0] = x0;
      Cf[(size_t)m * ldc + c1] = x1;
    }
  }
}

// ---------------------------------------------------------------------------
// Flash attention. One 16-row query tile per wave, 8 waves per block.
// K/V j-tiles (32 x HD bf16) staged block-wide into LDS via async DMA,
// double buffered: tile j+1 in flight while tile j feeds the WMMAs.
// MASK: 0=local |i-j|<=256, 1=local|(j%16==0), 2=global tokens, 3=none.
// ---------------------------------------------------------------------------
template <int HD, int MASK>
__global__ void flash_wmma(const unsigned short* __restrict__ Q,
                           const unsigned short* __restrict__ Kp,
                           const unsigned short* __restrict__ Vp,
                           int ldqkv, float* __restrict__ O, int ldo,
                           int S, int H, float scale)
{
  constexpr int NKC  = HD / 32;        // K chunks for scores
  constexpr int NOC  = HD / 16;        // output column tiles
  constexpr int CHNK = HD / 64;        // 16B async copies per thread per tile
  constexpr int AC   = 2 * CHNK;       // async instrs in flight per wave/tile

  __shared__ __align__(16) unsigned short Kt[2][32 * HD];
  __shared__ __align__(16) unsigned short Vt[2][32 * HD];
  __shared__ unsigned short Plds[8][16 * 32];
  __shared__ unsigned char  inG[2048];

  const int tid  = threadIdx.x;
  const int lane = tid & 31;
  const int wv   = tid >> 5;
  const int hi   = lane >> 4;
  const int nl   = lane & 15;
  const int b    = blockIdx.y / H;
  const int h    = blockIdx.y % H;
  const int i0   = (blockIdx.x * 8 + wv) * 16;
  const size_t rowbase = (size_t)b * S;

  if constexpr (MASK == 2) {
    for (int i = tid; i < 2048; i += blockDim.x) inG[i] = 0;
    __syncthreads();
    if (tid < 32) inG[(tid * (S - 1)) / 31] = 1;
    __syncthreads();
  }

  // stage K/V rows [j0, j0+32) of head h into buffer `buf`
  auto issue_tile = [&](int buf, int j0) {
#pragma unroll
    for (int c = 0; c < CHNK; ++c) {
      int e = (tid + c * 256) * 8;               // bf16 element index
      int j = e / HD, d = e % HD;
      size_t goff = (rowbase + j0 + j) * (size_t)ldqkv + h * HD + d;
      async_copy_b128(Kp + goff, &Kt[buf][j * HD + d]);
      async_copy_b128(Vp + goff, &Vt[buf][j * HD + d]);
    }
  };

  // Q fragments (A-layout), loaded once from global
  BfFrag qf[NKC];
  {
    const unsigned short* qrow = Q + (rowbase + i0 + nl) * ldqkv + h * HD;
#pragma unroll
    for (int c = 0; c < NKC; ++c)
#pragma unroll
      for (int v = 0; v < 8; ++v) {
        int g = v >> 2, v2 = v & 3;
        int k = c * 32 + g * 16 + hi * 8 + v2 * 2;
        qf[c].u[v] = *(const unsigned int*)&qrow[k];
      }
  }

  v8f oacc[NOC];
  float mrow[8], lrow[8];
#pragma unroll
  for (int oc = 0; oc < NOC; ++oc)
#pragma unroll
    for (int v = 0; v < 8; ++v) oacc[oc][v] = 0.f;
#pragma unroll
  for (int v = 0; v < 8; ++v) { mrow[v] = -1e30f; lrow[v] = 0.f; }

  issue_tile(0, 0);                     // prologue DMA

  for (int j0 = 0; j0 < S; j0 += 32) {
    const int cur = (j0 >> 5) & 1;
    if (j0 + 32 < S) {
      issue_tile(cur ^ 1, j0 + 32);     // next tile in flight
      asm volatile("s_wait_asynccnt %0" :: "i"(AC) : "memory");
    } else {
      asm volatile("s_wait_asynccnt 0" ::: "memory");
    }
    __syncthreads();                    // current tile resident block-wide

    const unsigned short* Kc = Kt[cur];
    const unsigned short* Vc = Vt[cur];

    v8f s0, s1;
#pragma unroll
    for (int v = 0; v < 8; ++v) { s0[v] = 0.f; s1[v] = 0.f; }

#pragma unroll
    for (int c = 0; c < NKC; ++c) {
      BfFrag kf;
      const unsigned short* k0r = &Kc[(size_t)nl * HD + c * 32];
#pragma unroll
      for (int v = 0; v < 8; ++v)
        kf.u[v] = *(const unsigned int*)&k0r[(v >> 2) * 16 + hi * 8 + (v & 3) * 2];
      s0 = wmma_bf16(qf[c], kf, s0);

      const unsigned short* k1r = &Kc[(size_t)(16 + nl) * HD + c * 32];
#pragma unroll
      for (int v = 0; v < 8; ++v)
        kf.u[v] = *(const unsigned int*)&k1r[(v >> 2) * 16 + hi * 8 + (v & 3) * 2];
      s1 = wmma_bf16(qf[c], kf, s1);
    }

    // mask + online softmax; row r = hi*8+v lives in 16 lanes of this half
    float p0[8], p1[8];
#pragma unroll
    for (int v = 0; v < 8; ++v) {
      int i  = i0 + hi * 8 + v;
      int jA = j0 + nl;
      int jB = j0 + 16 + nl;
      float a0 = s0[v] * scale;
      float a1 = s1[v] * scale;
      bool okA, okB;
      if constexpr (MASK == 0) {
        okA = abs(i - jA) <= 256; okB = abs(i - jB) <= 256;
      } else if constexpr (MASK == 1) {
        okA = (abs(i - jA) <= 256) || ((jA & 15) == 0);
        okB = (abs(i - jB) <= 256) || ((jB & 15) == 0);
      } else if constexpr (MASK == 2) {
        okA = inG[i] || inG[jA]; okB = inG[i] || inG[jB];
      } else {
        okA = true; okB = true;
      }
      a0 = okA ? a0 : -1e9f;
      a1 = okB ? a1 : -1e9f;

      float tm = fmaxf(a0, a1);
#pragma unroll
      for (int m = 1; m < 16; m <<= 1) tm = fmaxf(tm, __shfl_xor(tm, m, 32));
      float mnew = fmaxf(mrow[v], tm);
      float e0 = __expf(a0 - mnew);
      float e1 = __expf(a1 - mnew);
      float ts = e0 + e1;
#pragma unroll
      for (int m = 1; m < 16; m <<= 1) ts += __shfl_xor(ts, m, 32);
      float alpha = __expf(mrow[v] - mnew);
      lrow[v] = lrow[v] * alpha + ts;
      mrow[v] = mnew;
#pragma unroll
      for (int oc = 0; oc < NOC; ++oc) oacc[oc][v] *= alpha;
      p0[v] = e0; p1[v] = e1;
    }

    // P: C-layout -> LDS -> A-layout
    unsigned short* P = Plds[wv];
#pragma unroll
    for (int v = 0; v < 8; ++v) {
      int r = hi * 8 + v;
      P[r * 32 + nl]      = f2bf(p0[v]);
      P[r * 32 + 16 + nl] = f2bf(p1[v]);
    }
    BfFrag pf;
#pragma unroll
    for (int v = 0; v < 8; ++v) {
      int g = v >> 2, v2 = v & 3;
      int k = g * 16 + hi * 8 + v2 * 2;
      pf.u[v] = *(const unsigned int*)&P[nl * 32 + k];
    }

    // O += P(16x32) * V(32x16) per output column tile
#pragma unroll
    for (int oc = 0; oc < NOC; ++oc) {
      BfFrag vf;
#pragma unroll
      for (int v = 0; v < 8; ++v) {
        int k = (v >> 2) * 16 + hi * 8 + (v & 3) * 2; // j offset within tile
        const unsigned short* vb = &Vc[(size_t)k * HD + oc * 16 + nl];
        vf.u[v] = (unsigned int)vb[0] | ((unsigned int)vb[HD] << 16);
      }
      oacc[oc] = wmma_bf16(pf, vf, oacc[oc]);
    }
    __syncthreads();                    // before buffer reuse
  }

#pragma unroll
  for (int oc = 0; oc < NOC; ++oc)
#pragma unroll
    for (int v = 0; v < 8; ++v) {
      int i = i0 + hi * 8 + v;
      O[(rowbase + i) * ldo + h * HD + oc * 16 + nl] = oacc[oc][v] / lrow[v];
    }
}

// ---------------------------------------------------------------------------
// Small elementwise kernels
// ---------------------------------------------------------------------------
__global__ void pool_kernel(const float* __restrict__ x, float* __restrict__ pooled,
                            int S, int P, int D, int total)
{
  int idx = blockIdx.x * blockDim.x + threadIdx.x;
  if (idx >= total) return;
  int d = idx % D;
  int p = (idx / D) % P;
  int b = idx / (D * P);
  int r = S / P;
  const float* base = x + ((size_t)b * S + (size_t)p * r) * D + d;
  float s = 0.f;
  for (int t = 0; t < r; ++t) s += base[(size_t)t * D];
  pooled[idx] = s / (float)r;
}

__global__ void interp_pos_kernel(const float* __restrict__ projOut,
                                  const float* __restrict__ pos,
                                  float* __restrict__ feat,
                                  int S, int P, int D, int total)
{
  int idx = blockIdx.x * blockDim.x + threadIdx.x;
  if (idx >= total) return;
  int d = idx % D;
  int s = (idx / D) % S;
  int b = idx / (D * S);
  float t = ((s + 0.5f) * (float)P) / (float)S - 0.5f;
  t = fminf(fmaxf(t, 0.f), (float)(P - 1));
  int i0 = (int)t;
  float f = t - (float)i0;
  int i1 = min(i0 + 1, P - 1);
  float v0 = projOut[((size_t)b * P + i0) * D + d] + pos[(size_t)i0 * D + d];
  float v1 = projOut[((size_t)b * P + i1) * D + d] + pos[(size_t)i1 * D + d];
  feat[idx] = v0 + f * (v1 - v0);
}

__global__ void coupling_kernel(const float* __restrict__ feat,
                                const float* __restrict__ scalef,
                                float* __restrict__ cout, int n, int D)
{
  int idx = blockIdx.x * blockDim.x + threadIdx.x;
  if (idx >= n) return;
  int d = idx % D;
  size_t row = idx / D;
  float s = 0.f;
#pragma unroll
  for (int l = 0; l < 3; ++l)
    s += feat[(size_t)l * n + idx] + scalef[row * (3 * D) + l * D + d];
  cout[idx] = s * (1.f / 3.f);
}

__global__ void resid_ln_kernel(const float* __restrict__ x,
                                const float* __restrict__ t,
                                const float* __restrict__ g,
                                const float* __restrict__ be,
                                float* __restrict__ out, int D)
{
  __shared__ float sbuf[256];
  const int row = blockIdx.x;
  const int tid = threadIdx.x;
  const size_t base = (size_t)row * D;
  float v0 = x[base + tid]       + t[base + tid];
  float v1 = x[base + 256 + tid] + t[base + 256 + tid];

  sbuf[tid] = v0 + v1;
  __syncthreads();
  for (int off = 128; off > 0; off >>= 1) {
    if (tid < off) sbuf[tid] += sbuf[tid + off];
    __syncthreads();
  }
  float mu = sbuf[0] / (float)D;
  __syncthreads();

  float d0 = v0 - mu, d1 = v1 - mu;
  sbuf[tid] = d0 * d0 + d1 * d1;
  __syncthreads();
  for (int off = 128; off > 0; off >>= 1) {
    if (tid < off) sbuf[tid] += sbuf[tid + off];
    __syncthreads();
  }
  float rstd = rsqrtf(sbuf[0] / (float)D + 1e-5f);
  out[base + tid]       = d0 * rstd * g[tid]       + be[tid];
  out[base + 256 + tid] = d1 * rstd * g[256 + tid] + be[256 + tid];
}

// ---------------------------------------------------------------------------
// Orchestration
// ---------------------------------------------------------------------------
extern "C" void kernel_launch(void* const* d_in, const int* in_sizes, int n_in,
                              void* d_out, int out_size, void* d_ws, size_t ws_size,
                              hipStream_t stream)
{
  (void)in_sizes; (void)n_in; (void)out_size; (void)ws_size;

  const float* x      = (const float*)d_in[0];
  const float* proj_w = (const float*)d_in[1];
  const float* proj_b = (const float*)d_in[2];
  const float* pos    = (const float*)d_in[3];
  const float* wq = (const float*)d_in[4];  const float* bq = (const float*)d_in[5];
  const float* wk = (const float*)d_in[6];  const float* bk = (const float*)d_in[7];
  const float* wvv= (const float*)d_in[8];  const float* bv = (const float*)d_in[9];
  const float* wo = (const float*)d_in[10]; const float* bo = (const float*)d_in[11];
  const float* fiw= (const float*)d_in[12]; const float* fib= (const float*)d_in[13];
  const float* fow= (const float*)d_in[14]; const float* fob= (const float*)d_in[15];
  const float* opw= (const float*)d_in[16]; const float* opb= (const float*)d_in[17];
  const float* lng= (const float*)d_in[18]; const float* lnb= (const float*)d_in[19];
  float* out = (float*)d_out;

  constexpr int B = 2, S = 2048, D = 512, H = 8, D3 = 1536, N4 = 4608;
  constexpr int BS = B * S;
  const int Pl[3] = {512, 128, 32};
  const size_t nBSD = (size_t)BS * D;

  float* feat   = (float*)d_ws;                              // 3 * BS*D
  float* pooled = feat + 3 * nBSD;
  float* featP  = pooled + (size_t)B * 512 * D;
  unsigned short* qkvB = (unsigned short*)(featP + (size_t)B * 512 * D);
  float* attnO  = (float*)((char*)qkvB + (size_t)BS * N4 * sizeof(unsigned short));
  float* scalef = attnO + (size_t)BS * D3;
  float* tmpOut = scalef + (size_t)BS * D3;
  float* fusedB = feat;  // feats reusable once coupling is written

  auto gemm = [&](const float* A, int lda, const float* W, int ldw,
                  const float* bias, float* Cf, unsigned short* Cb,
                  int ldc, int M, int N, int K) {
    dim3 grid(N / 64, M / 64);
    gemm_bf16_wmma<<<grid, 256, 0, stream>>>(A, lda, W, ldw, bias, Cf, Cb,
                                             ldc, M, N, K);
  };

  const dim3 fgrid(S / 128, B * H);
  const float sc64  = 0.125f;
  const float sc192 = 0.0721687836487032f;

  for (int l = 0; l < 3; ++l) {
    const int P = Pl[l];
    const int totP = B * P * D;
    pool_kernel<<<(totP + 255) / 256, 256, 0, stream>>>(x, pooled, S, P, D, totP);

    gemm(pooled, D, proj_w + (size_t)l * D * D, D, proj_b + (size_t)l * D,
         featP, nullptr, D, B * P, D, D);

    interp_pos_kernel<<<(int)(nBSD / 256), 256, 0, stream>>>(
        featP, pos + (size_t)l * 10000 * D, feat + (size_t)l * nBSD,
        S, P, D, (int)nBSD);

    const float* fl = feat + (size_t)l * nBSD;
    gemm(fl, D, wq  + (size_t)l * D * D, D, bq + (size_t)l * D,
         nullptr, qkvB + 0,    N4, BS, D, D);
    gemm(fl, D, wk  + (size_t)l * D * D, D, bk + (size_t)l * D,
         nullptr, qkvB + D3,   N4, BS, D, D);
    gemm(fl, D, wvv + (size_t)l * D * D, D, bv + (size_t)l * D,
         nullptr, qkvB + 2*D3, N4, BS, D, D);

    if (l == 0)
      flash_wmma<64, 0><<<fgrid, 256, 0, stream>>>(qkvB, qkvB + D3, qkvB + 2*D3,
                                                   N4, attnO, D, S, H, sc64);
    else if (l == 1)
      flash_wmma<64, 1><<<fgrid, 256, 0, stream>>>(qkvB, qkvB + D3, qkvB + 2*D3,
                                                   N4, attnO, D, S, H, sc64);
    else
      flash_wmma<64, 2><<<fgrid, 256, 0, stream>>>(qkvB, qkvB + D3, qkvB + 2*D3,
                                                   N4, attnO, D, S, H, sc64);

    gemm(attnO, D, wo + (size_t)l * D * D, D, bo + (size_t)l * D,
         scalef + (size_t)l * D, nullptr, D3, BS, D, D);
  }

  coupling_kernel<<<(int)(nBSD / 256), 256, 0, stream>>>(
      feat, scalef, out + nBSD, (int)nBSD, D);

  gemm(scalef, D3, fiw, D3, fib, nullptr, qkvB, N4, BS, N4, D3);

  flash_wmma<192, 3><<<fgrid, 256, 0, stream>>>(qkvB, qkvB + D3, qkvB + 2*D3,
                                                N4, attnO, D3, S, H, sc192);

  gemm(attnO, D3, fow, D3, fob, fusedB, nullptr, D3, BS, D3, D3);
  gemm(fusedB, D3, opw, D3, opb, tmpOut, nullptr, D, BS, D, D3);

  resid_ln_kernel<<<BS, 256, 0, stream>>>(x, tmpOut, lng, lnb, out, D);
}
